// GraphModel_70652212019572
// MI455X (gfx1250) — compile-verified, matrix operand order checked by reading, not run on hardware
//
#include <hip/hip_runtime.h>
#include <hip/hip_bf16.h>

typedef __attribute__((ext_vector_type(16))) _Float16 v16h;
typedef __attribute__((ext_vector_type(2)))  _Float16 h2f;
typedef __attribute__((ext_vector_type(8)))  float    v8f;

#define N_TOK 512
#define WD    300
#define PD    100
#define IN_D  400
#define KPAD  416     // 400 padded to multiple of 32 for WMMA K
#define GATES 2048    // 4*H
#define HHID  512     // per-direction hidden
#define LHID  1024    // bidirectional hidden
#define HID   256     // mlp hidden
#define NRL   50

#define WMMA_F16(A_, B_, C_) \
  __builtin_amdgcn_wmma_f32_16x16x32_f16(false, (A_), false, (B_), (short)0, (C_), false, false)

// -------------------------------------------------------------------------
// Packing / conversion kernels
// -------------------------------------------------------------------------

// x = concat(Wemb[word_ids], Pemb[pos_ids]) -> f16, zero-padded to KPAD
__global__ void pack_x(const int* __restrict__ wid, const int* __restrict__ pid,
                       const float* __restrict__ Wemb, const float* __restrict__ Pemb,
                       _Float16* __restrict__ xh) {
  int idx = blockIdx.x * blockDim.x + threadIdx.x;
  if (idx >= N_TOK * KPAD) return;
  int pos = idx / KPAD, k = idx % KPAD;
  float v = 0.f;
  if (k < WD)        v = Wemb[(size_t)wid[pos] * WD + k];
  else if (k < IN_D) v = Pemb[(size_t)pid[pos] * PD + (k - WD)];
  xh[idx] = (_Float16)v;
}

// generic f32[rows][kin] -> f16[rows][kpad] zero-padded
__global__ void pack_pad_f16(const float* __restrict__ src, _Float16* __restrict__ dst,
                             int rows, int kin, int kpad) {
  int idx = blockIdx.x * blockDim.x + threadIdx.x;
  if (idx >= rows * kpad) return;
  int r = idx / kpad, k = idx % kpad;
  dst[idx] = (_Float16)(k < kin ? src[(size_t)r * kin + k] : 0.f);
}

// Whh f32 [GATES][HHID] -> transposed half2 [HHID/2][GATES]:
//   W2[kk][g] = { Whh[g][2kk], Whh[g][2kk+1] }  (coalesced recurrent matvec reads)
__global__ void pack_whh2(const float* __restrict__ whh, h2f* __restrict__ w2) {
  int idx = blockIdx.x * blockDim.x + threadIdx.x;
  if (idx >= (HHID / 2) * GATES) return;
  int kk = idx / GATES, g = idx % GATES;
  h2f v;
  v.x = (_Float16)whh[(size_t)g * HHID + 2 * kk];
  v.y = (_Float16)whh[(size_t)g * HHID + 2 * kk + 1];
  w2[idx] = v;
}

// -------------------------------------------------------------------------
// WMMA GEMM:  C[M][N] = A[M][K] @ B[N][K]^T + bias[N]
// A,B f16 row-major, K multiple of 32 (>= 64), M mult of 16, N mult of 64.
// One wave computes a 16x64 strip: 4 accumulators share one A fragment.
// Software-pipelined: next iteration's A + 4xB loads are issued before the
// current 4 WMMAs consume their fragments (double-buffered registers), so
// loads overlap matrix ops instead of draining loadcnt to 0 per WMMA.
// Grid sized so waves == tiles (EXEC all-1s for every WMMA).
// -------------------------------------------------------------------------
__global__ __launch_bounds__(256) void wmma_gemm4(
    const _Float16* __restrict__ A, const _Float16* __restrict__ B,
    const float* __restrict__ bias, float* __restrict__ C,
    int M, int N, int K) {
  int wave = (int)((blockIdx.x * blockDim.x + threadIdx.x) >> 5);
  int lane = threadIdx.x & 31;
  int tilesN = N >> 6;                     // 64-wide N tiles
  int tiles  = (M >> 4) * tilesN;
  if (wave >= tiles) return;               // wave-uniform
  int tm = wave / tilesN, tn = wave % tilesN;

  // A fragment: lane L -> row L%16, K-slice (L/16)*16 .. +15 (32B contiguous)
  const _Float16* arow = A + (size_t)(tm * 16 + (lane & 15)) * K + ((lane >> 4) * 16);
  // B fragments: identical lane pattern on the [N][K] weight matrix, 4 strips
  const _Float16* b0 = B + (size_t)(tn * 64 + (lane & 15)) * K + ((lane >> 4) * 16);
  const _Float16* b1 = b0 + (size_t)16 * K;
  const _Float16* b2 = b0 + (size_t)32 * K;
  const _Float16* b3 = b0 + (size_t)48 * K;

  v8f acc0 = {}, acc1 = {}, acc2 = {}, acc3 = {};

  // prologue: load first fragment set
  v16h a   = *(const v16h*)(arow);
  v16h vb0 = *(const v16h*)(b0);
  v16h vb1 = *(const v16h*)(b1);
  v16h vb2 = *(const v16h*)(b2);
  v16h vb3 = *(const v16h*)(b3);

  for (int k0 = 32; k0 < K; k0 += 32) {
    // issue next-iteration loads before consuming current fragments
    v16h an  = *(const v16h*)(arow + k0);
    v16h nb0 = *(const v16h*)(b0 + k0);
    v16h nb1 = *(const v16h*)(b1 + k0);
    v16h nb2 = *(const v16h*)(b2 + k0);
    v16h nb3 = *(const v16h*)(b3 + k0);

    acc0 = WMMA_F16(a, vb0, acc0);
    acc1 = WMMA_F16(a, vb1, acc1);
    acc2 = WMMA_F16(a, vb2, acc2);
    acc3 = WMMA_F16(a, vb3, acc3);

    a = an; vb0 = nb0; vb1 = nb1; vb2 = nb2; vb3 = nb3;
  }

  // epilogue: consume last fragment set
  acc0 = WMMA_F16(a, vb0, acc0);
  acc1 = WMMA_F16(a, vb1, acc1);
  acc2 = WMMA_F16(a, vb2, acc2);
  acc3 = WMMA_F16(a, vb3, acc3);

  // D layout: lane L reg r -> (row = r + 8*(L/16), col = L%16)
  int row0 = tm * 16 + (lane >> 4) * 8;
  int c0 = tn * 64 + (lane & 15);
  float bv0 = bias ? bias[c0]      : 0.f;
  float bv1 = bias ? bias[c0 + 16] : 0.f;
  float bv2 = bias ? bias[c0 + 32] : 0.f;
  float bv3 = bias ? bias[c0 + 48] : 0.f;
#pragma unroll
  for (int r = 0; r < 8; ++r) {
    float* crow = C + (size_t)(row0 + r) * N + c0;
    crow[0]  = acc0[r] + bv0;
    crow[16] = acc1[r] + bv1;
    crow[32] = acc2[r] + bv2;
    crow[48] = acc3[r] + bv3;
  }
}

// -------------------------------------------------------------------------
// Sequential LSTM: grid = {fwd, bwd}, 512 threads; thread j owns hidden unit j.
// pre-gates (x@Wih^T + b) already computed; recurrent matvec from LDS h.
// -------------------------------------------------------------------------
__device__ __forceinline__ float sigmf(float x) { return 1.f / (1.f + __expf(-x)); }

__global__ __launch_bounds__(512) void lstm_kernel(
    const float* __restrict__ preF, const float* __restrict__ preB,
    const h2f* __restrict__ W2F, const h2f* __restrict__ W2B,
    _Float16* __restrict__ lstmH) {
  __shared__ float hs[HHID];
  int j   = threadIdx.x;
  int dir = blockIdx.x;
  const float* pre = dir ? preB : preF;
  const h2f*   W2  = dir ? W2B  : W2F;

  float c = 0.f;
  hs[j] = 0.f;
  __syncthreads();

  for (int t = 0; t < N_TOK; ++t) {
    int pos = dir ? (N_TOK - 1 - t) : t;
    const float* pg = pre + (size_t)pos * GATES;
    float ai = pg[j];
    float af = pg[HHID + j];
    float ag = pg[2 * HHID + j];
    float ao = pg[3 * HHID + j];

#pragma unroll 4
    for (int kk = 0; kk < HHID / 2; ++kk) {
      float2 hv = ((const float2*)hs)[kk];
      const h2f* wrow = W2 + (size_t)kk * GATES;
      h2f wi = wrow[j];
      h2f wf = wrow[HHID + j];
      h2f wg = wrow[2 * HHID + j];
      h2f wo = wrow[3 * HHID + j];
      ai += hv.x * (float)wi.x + hv.y * (float)wi.y;
      af += hv.x * (float)wf.x + hv.y * (float)wf.y;
      ag += hv.x * (float)wg.x + hv.y * (float)wg.y;
      ao += hv.x * (float)wo.x + hv.y * (float)wo.y;
    }

    float ig = sigmf(ai), fg = sigmf(af), gg = tanhf(ag), og = sigmf(ao);
    c = fg * c + ig * gg;
    float h = og * tanhf(c);

    __syncthreads();               // all reads of previous h done
    hs[j] = h;
    lstmH[(size_t)pos * LHID + dir * HHID + j] = (_Float16)h;
    __syncthreads();               // new h visible
  }
}

// -------------------------------------------------------------------------
// Pair scoring: scores[i][j] = out_b + sum_k out_w[k] * tanh(hh[i][k]+tt[j][k])
// 16x16 output tile per 256-thread block; LDS-tiled, rows padded (no conflicts).
// -------------------------------------------------------------------------
__global__ __launch_bounds__(256) void pair_kernel(
    const float* __restrict__ hh, const float* __restrict__ tt,
    const float* __restrict__ ow, const float* __restrict__ ob,
    float* __restrict__ out) {
  __shared__ float sh[16][HID + 1];
  __shared__ float st[16][HID + 1];
  __shared__ float sw[HID];
  int tid = threadIdx.x;
  int bi = blockIdx.y * 16, bj = blockIdx.x * 16;
  for (int idx = tid; idx < 16 * HID; idx += 256) {
    int r = idx / HID, k = idx % HID;
    sh[r][k] = hh[(size_t)(bi + r) * HID + k];
    st[r][k] = tt[(size_t)(bj + r) * HID + k];
  }
  for (int k = tid; k < HID; k += 256) sw[k] = ow[k];
  __syncthreads();

  int ti = tid >> 4, tj = tid & 15;
  float s = ob[0];
#pragma unroll 4
  for (int k = 0; k < HID; ++k)
    s += sw[k] * tanhf(sh[ti][k] + st[tj][k]);
  out[(size_t)(bi + ti) * N_TOK + (bj + tj)] = s;
}

// -------------------------------------------------------------------------
// Relation scoring at gold heads: one block per token i+1.
// -------------------------------------------------------------------------
__global__ __launch_bounds__(64) void rel_kernel(
    const float* __restrict__ rta, const float* __restrict__ rha,
    const int* __restrict__ heads, const float* __restrict__ rw,
    const float* __restrict__ rb, float* __restrict__ out) {
  __shared__ float rep[HID];
  int i = blockIdx.x;                 // token index i+1
  int head = heads[i + 1];
  for (int k = threadIdx.x; k < HID; k += 64)
    rep[k] = tanhf(rta[(size_t)(i + 1) * HID + k] + rha[(size_t)head * HID + k]);
  __syncthreads();
  for (int r = threadIdx.x; r < NRL; r += 64) {
    float s = rb[r];
    for (int k = 0; k < HID; ++k) s += rep[k] * rw[(size_t)r * HID + k];
    out[(size_t)i * NRL + r] = s;
  }
}

// -------------------------------------------------------------------------
// Host-side launch
// -------------------------------------------------------------------------
static inline size_t align256(size_t x) { return (x + 255) & ~(size_t)255; }

extern "C" void kernel_launch(void* const* d_in, const int* in_sizes, int n_in,
                              void* d_out, int out_size, void* d_ws, size_t ws_size,
                              hipStream_t stream) {
  (void)in_sizes; (void)n_in; (void)out_size; (void)ws_size;

  const int*   word_ids = (const int*)d_in[0];
  const int*   pos_ids  = (const int*)d_in[1];
  const int*   heads    = (const int*)d_in[2];
  const float* Wemb     = (const float*)d_in[3];
  const float* Pemb     = (const float*)d_in[4];
  const float* Wih_f    = (const float*)d_in[5];
  const float* Whh_f    = (const float*)d_in[6];
  const float* b_f      = (const float*)d_in[7];
  const float* Wih_b    = (const float*)d_in[8];
  const float* Whh_b    = (const float*)d_in[9];
  const float* b_b      = (const float*)d_in[10];
  const float* hmlp_w   = (const float*)d_in[11];
  const float* hmlp_b   = (const float*)d_in[12];
  const float* tmlp_w   = (const float*)d_in[13];
  const float* tmlp_b   = (const float*)d_in[14];
  const float* out_w    = (const float*)d_in[15];
  const float* out_b    = (const float*)d_in[16];
  const float* rh_w     = (const float*)d_in[17];
  const float* rh_b     = (const float*)d_in[18];
  const float* rt_w     = (const float*)d_in[19];
  const float* rt_b     = (const float*)d_in[20];
  const float* rout_w   = (const float*)d_in[21];
  const float* rout_b   = (const float*)d_in[22];

  // ---- workspace carve-up (all offsets 256B aligned) ----
  char* ws = (char*)d_ws;
  size_t off = 0;
  auto carve = [&](size_t bytes) { char* p = ws + off; off += align256(bytes); return p; };

  _Float16* XH    = (_Float16*)carve((size_t)N_TOK * KPAD * 2);
  _Float16* WIHF  = (_Float16*)carve((size_t)GATES * KPAD * 2);
  _Float16* WIHB  = (_Float16*)carve((size_t)GATES * KPAD * 2);
  float*    PREF  = (float*)   carve((size_t)N_TOK * GATES * 4);
  float*    PREB  = (float*)   carve((size_t)N_TOK * GATES * 4);
  h2f*      W2F   = (h2f*)     carve((size_t)(HHID/2) * GATES * 4);
  h2f*      W2B   = (h2f*)     carve((size_t)(HHID/2) * GATES * 4);
  _Float16* LSTMH = (_Float16*)carve((size_t)N_TOK * LHID * 2);
  _Float16* HMW   = (_Float16*)carve((size_t)HID * LHID * 2);
  _Float16* TMW   = (_Float16*)carve((size_t)HID * LHID * 2);
  _Float16* RHW   = (_Float16*)carve((size_t)HID * LHID * 2);
  _Float16* RTW   = (_Float16*)carve((size_t)HID * LHID * 2);
  float*    HH    = (float*)   carve((size_t)N_TOK * HID * 4);
  float*    TT    = (float*)   carve((size_t)N_TOK * HID * 4);
  float*    RHA   = (float*)   carve((size_t)N_TOK * HID * 4);
  float*    RTA   = (float*)   carve((size_t)N_TOK * HID * 4);

  float* head_scores = (float*)d_out;
  float* r_scores    = (float*)d_out + (size_t)N_TOK * N_TOK;

  // ---- packing ----
  {
    int tot = N_TOK * KPAD;
    pack_x<<<(tot + 255) / 256, 256, 0, stream>>>(word_ids, pos_ids, Wemb, Pemb, XH);
  }
  {
    int tot = GATES * KPAD;
    pack_pad_f16<<<(tot + 255) / 256, 256, 0, stream>>>(Wih_f, WIHF, GATES, IN_D, KPAD);
    pack_pad_f16<<<(tot + 255) / 256, 256, 0, stream>>>(Wih_b, WIHB, GATES, IN_D, KPAD);
  }
  {
    int tot = HID * LHID;
    pack_pad_f16<<<(tot + 255) / 256, 256, 0, stream>>>(hmlp_w, HMW, HID, LHID, LHID);
    pack_pad_f16<<<(tot + 255) / 256, 256, 0, stream>>>(tmlp_w, TMW, HID, LHID, LHID);
    pack_pad_f16<<<(tot + 255) / 256, 256, 0, stream>>>(rh_w,   RHW, HID, LHID, LHID);
    pack_pad_f16<<<(tot + 255) / 256, 256, 0, stream>>>(rt_w,   RTW, HID, LHID, LHID);
  }
  {
    int tot = (HHID / 2) * GATES;
    pack_whh2<<<(tot + 255) / 256, 256, 0, stream>>>(Whh_f, W2F);
    pack_whh2<<<(tot + 255) / 256, 256, 0, stream>>>(Whh_b, W2B);
  }

  // ---- input-projection pre-gates: WMMA GEMM [512,416] @ [2048,416]^T ----
  {
    int waves = (N_TOK / 16) * (GATES / 64);      // 32*32 = 1024 waves, 8/block
    wmma_gemm4<<<waves / 8, 256, 0, stream>>>(XH, WIHF, b_f, PREF, N_TOK, GATES, KPAD);
    wmma_gemm4<<<waves / 8, 256, 0, stream>>>(XH, WIHB, b_b, PREB, N_TOK, GATES, KPAD);
  }

  // ---- sequential BiLSTM (latency-bound critical path) ----
  lstm_kernel<<<2, 512, 0, stream>>>(PREF, PREB, W2F, W2B, LSTMH);

  // ---- 4 projections: WMMA GEMM [512,1024] @ [256,1024]^T ----
  {
    int waves = (N_TOK / 16) * (HID / 64);        // 32*4 = 128 waves, 8/block
    wmma_gemm4<<<waves / 8, 256, 0, stream>>>(LSTMH, HMW, hmlp_b, HH,  N_TOK, HID, LHID);
    wmma_gemm4<<<waves / 8, 256, 0, stream>>>(LSTMH, TMW, tmlp_b, TT,  N_TOK, HID, LHID);
    wmma_gemm4<<<waves / 8, 256, 0, stream>>>(LSTMH, RHW, rh_b,   RHA, N_TOK, HID, LHID);
    wmma_gemm4<<<waves / 8, 256, 0, stream>>>(LSTMH, RTW, rt_b,   RTA, N_TOK, HID, LHID);
  }

  // ---- fused all-pairs head scoring (tanh-bound) ----
  {
    dim3 grid(N_TOK / 16, N_TOK / 16);
    pair_kernel<<<grid, 256, 0, stream>>>(HH, TT, out_w, out_b, head_scores);
  }

  // ---- relation scoring at gold heads ----
  rel_kernel<<<N_TOK - 1, 64, 0, stream>>>(RTA, RHA, heads, rout_w, rout_b, r_scores);
}
